// DecoderBlock_8272107012286
// MI455X (gfx1250) — matrix operand order repaired and checked by reading, not compiled, not run on hardware
//
#include <hip/hip_runtime.h>
#include <cstdint>

#define B_  2
#define S_  2048
#define D_  1024
#define H_  16
#define DK_ 64
#define DF_ 4096
#define M_  (B_*S_)   // 4096 total token rows

typedef __bf16 bf16;
typedef __attribute__((ext_vector_type(16))) __bf16 v16bf;
typedef __attribute__((ext_vector_type(8)))  float  v8f;

struct Frag { uint4 lo, hi; };           // 32 bytes = 16 bf16 (one WMMA operand per lane)
struct BF4  { bf16 a, b, c, d; };        // 8 bytes

static __device__ __forceinline__ v16bf asv(Frag f) { return __builtin_bit_cast(v16bf, f); }

static __device__ __forceinline__ bf16 f2bf(float f) {
  uint32_t u = __builtin_bit_cast(uint32_t, f);
  u += 0x7FFFu + ((u >> 16) & 1u);       // round-to-nearest-even
  uint16_t s = (uint16_t)(u >> 16);
  return __builtin_bit_cast(bf16, s);
}
static __device__ __forceinline__ float bf2f(bf16 b) {
  uint16_t s = __builtin_bit_cast(uint16_t, b);
  uint32_t u = ((uint32_t)s) << 16;
  return __builtin_bit_cast(float, u);
}
static __device__ __forceinline__ v8f vzero() {
  v8f z = {0.f,0.f,0.f,0.f,0.f,0.f,0.f,0.f}; return z;
}
static __device__ __forceinline__ float gelu_exact(float x) {
  return 0.5f * x * (1.f + erff(x * 0.70710678118654752f));
}
static __device__ __forceinline__ v8f wmma_bf16(Frag a, Frag b, v8f c) {
  // D(16x16 f32) = A(16x32 bf16) * B(32x16 bf16) + C
  return __builtin_amdgcn_wmma_f32_16x16x32_bf16(false, asv(a), false, asv(b),
                                                 (short)0, c, false, false);
}

// ---- CDNA5 async global->LDS DMA (ASYNCcnt-tracked) via inline asm ----------
// ISA 15.18.3 opcode 98: copies global memory at vaddr64 into LDS at the per-lane VDST address plus offset; the VDST VGPR
// holds the per-lane LDS byte address == low 32 bits of a generic LDS pointer.
#define USE_ASYNC_LDS 1
#if USE_ASYNC_LDS
static __device__ __forceinline__ void async_ld128(const void* g, void* l) {
  unsigned lds32 = (unsigned)(size_t)l;
  unsigned long long ga = (unsigned long long)(size_t)g;
  asm volatile("global_load_async_to_lds_b128 %0, %1, off"
               :: "v"(lds32), "v"(ga) : "memory");
}
static __device__ __forceinline__ void async_wait0() {
  asm volatile("s_wait_asynccnt 0" ::: "memory");
}
#endif

// ---------------------------------------------------------------- cvt f32->bf16
__global__ void cvt_f32_to_bf16_kernel(const float* __restrict__ src,
                                       bf16* __restrict__ dst, int n4) {
  int i = blockIdx.x * blockDim.x + threadIdx.x;
  if (i >= n4) return;
  float4 v = reinterpret_cast<const float4*>(src)[i];
  BF4 o = { f2bf(v.x), f2bf(v.y), f2bf(v.z), f2bf(v.w) };
  reinterpret_cast<uint2*>(dst)[i] = __builtin_bit_cast(uint2, o);
}

// ---------------------------------------------------------------- RMSNorm -> bf16
// one block (256 threads) per row of D_=1024 floats
__global__ void rmsnorm_bf16_kernel(const float* __restrict__ x,
                                    const float* __restrict__ w,
                                    bf16* __restrict__ out) {
  int row = blockIdx.x;
  int tid = threadIdx.x;
  const float4 v = reinterpret_cast<const float4*>(x + (size_t)row * D_)[tid];
  float ss = v.x*v.x + v.y*v.y + v.z*v.z + v.w*v.w;
#pragma unroll
  for (int m = 1; m < 32; m <<= 1) ss += __shfl_xor(ss, m, 32);
  __shared__ float red[8];
  if ((tid & 31) == 0) red[tid >> 5] = ss;
  __syncthreads();
  float tot = 0.f;
#pragma unroll
  for (int i = 0; i < 8; ++i) tot += red[i];
  float r = rsqrtf(tot * (1.f / (float)D_) + 1e-6f);
  const float4 wv = reinterpret_cast<const float4*>(w)[tid];
  BF4 o = { f2bf(v.x*r*wv.x), f2bf(v.y*r*wv.y), f2bf(v.z*r*wv.z), f2bf(v.w*r*wv.w) };
  reinterpret_cast<uint2*>(out + (size_t)row * D_)[tid] = __builtin_bit_cast(uint2, o);
}

// ---------------------------------------------------------------- WMMA GEMM
// C[M,N] = act( A[M,K](bf16) @ W[N,K]^T(bf16) [+ bias] [+ residual] )
// GATED==1: out_bf16 = gelu(A@W^T) * (A@Wg^T)   (SwiGLU gate, no bias/residual)
// block = 256 thr (8 waves), tile BM=128 x BN=64, K step 32.
// Ping-pong double buffering: async global->LDS DMA for tile t+1 overlaps the
// WMMA compute on tile t; s_wait_asynccnt 0 lands after the math.
#define LDA 40   // 32 + 8 bf16 pad
template <int GATED>
__global__ void gemm_wmma_kernel(const bf16* __restrict__ A,
                                 const bf16* __restrict__ W,
                                 const bf16* __restrict__ Wg,
                                 const float* __restrict__ bias,
                                 const float* __restrict__ residual,
                                 float* __restrict__ outF,
                                 bf16*  __restrict__ outB,
                                 int N, int K) {
  __shared__ bf16 ldsA[2][128 * LDA];
  __shared__ bf16 ldsW[2][64 * LDA];
  __shared__ bf16 ldsWg[GATED ? 2 * 64 * LDA : 8];
  const int tid  = threadIdx.x;
  const int wave = tid >> 5;
  const int lane = tid & 31;
  const int lh   = lane >> 4;     // half-wave select
  const int ll   = lane & 15;
  const int m0   = blockIdx.y * 128 + wave * 16;
  const int n0   = blockIdx.x * 64;

  // staging coordinates
  const int ar = tid >> 1;              // A row 0..127
  const int ac = (tid & 1) * 16;        // A col chunk 0|16
  const int wr = tid >> 2;              // W row 0..63
  const int wc = (tid & 3) * 8;         // W col chunk 0|8|16|24

  auto stage = [&](int p, int k0) {
    const bf16* srcA = A + (size_t)(blockIdx.y * 128 + ar) * K + k0 + ac;
    bf16*       dstA = &ldsA[p][ar * LDA + ac];
    const bf16* srcW = W + (size_t)(n0 + wr) * K + k0 + wc;
    bf16*       dstW = &ldsW[p][wr * LDA + wc];
#if USE_ASYNC_LDS
    async_ld128(srcA,     dstA);
    async_ld128(srcA + 8, dstA + 8);
    async_ld128(srcW,     dstW);
    if (GATED) {
      const bf16* srcG = Wg + (size_t)(n0 + wr) * K + k0 + wc;
      async_ld128(srcG, &ldsWg[p * 64 * LDA + wr * LDA + wc]);
    }
#else
    const uint4* sa = reinterpret_cast<const uint4*>(srcA);
    uint4*       da = reinterpret_cast<uint4*>(dstA);
    da[0] = sa[0];
    da[1] = sa[1];
    *reinterpret_cast<uint4*>(dstW) = *reinterpret_cast<const uint4*>(srcW);
    if (GATED) {
      *reinterpret_cast<uint4*>(&ldsWg[p * 64 * LDA + wr * LDA + wc]) =
          *reinterpret_cast<const uint4*>(Wg + (size_t)(n0 + wr) * K + k0 + wc);
    }
#endif
  };

  v8f acc[4]  = {vzero(), vzero(), vzero(), vzero()};
  v8f acc2[4] = {vzero(), vzero(), vzero(), vzero()};

  stage(0, 0);
#if USE_ASYNC_LDS
  async_wait0();
#endif
  __syncthreads();

  int p = 0;
  for (int k0 = 0; k0 < K; k0 += 32) {
    if (k0 + 32 < K) {
      stage(p ^ 1, k0 + 32);              // DMA next tile while computing this one
      __builtin_prefetch(A + (size_t)(blockIdx.y * 128 + ar) * K + k0 + 64 + ac, 0, 0);
    }

    Frag a;   // A 16x32: lane row = ll; lanes<16 K=0..15, lanes>=16 K=16..31
    {
      const bf16* pa_ = &ldsA[p][(wave * 16 + ll) * LDA + lh * 16];
      a.lo = *reinterpret_cast<const uint4*>(pa_);
      a.hi = *reinterpret_cast<const uint4*>(pa_ + 8);
    }
#pragma unroll
    for (int nt = 0; nt < 4; ++nt) {
      // B 32x16: lane col = ll; lanes<16 K=0..15, lanes>=16 K=16..31 -> contiguous row
      const bf16* pb = &ldsW[p][(nt * 16 + ll) * LDA + lh * 16];
      Frag b; b.lo = *reinterpret_cast<const uint4*>(pb);
              b.hi = *reinterpret_cast<const uint4*>(pb + 8);
      acc[nt] = wmma_bf16(a, b, acc[nt]);
      if (GATED) {
        const bf16* pg = &ldsWg[p * 64 * LDA + (nt * 16 + ll) * LDA + lh * 16];
        Frag g; g.lo = *reinterpret_cast<const uint4*>(pg);
                g.hi = *reinterpret_cast<const uint4*>(pg + 8);
        acc2[nt] = wmma_bf16(a, g, acc2[nt]);
      }
    }

#if USE_ASYNC_LDS
    async_wait0();                        // next-tile DMA done (ran under the WMMAs)
#endif
    __syncthreads();
    p ^= 1;
  }

  // epilogue: C layout -> lane col n = n0+nt*16+ll ; VGPR r -> row m0 + r + 8*lh
#pragma unroll
  for (int nt = 0; nt < 4; ++nt) {
    int n = n0 + nt * 16 + ll;
    float bval = bias ? bias[n] : 0.f;
#pragma unroll
    for (int r = 0; r < 8; ++r) {
      int m = m0 + r + lh * 8;
      size_t idx = (size_t)m * N + n;
      float val;
      if (GATED) {
        val = gelu_exact(acc[nt][r]) * acc2[nt][r];
      } else {
        val = acc[nt][r] + bval;
        if (residual) val += residual[idx];
      }
      if (outF) outF[idx] = val;
      if (outB) outB[idx] = f2bf(val);
    }
  }
}

// ---------------------------------------------------------------- RoPE (in place on q,k)
// layout [B,S,H,DK]; one thread per even/odd pair
__global__ void rope_kernel(bf16* __restrict__ q, bf16* __restrict__ k) {
  int idx = blockIdx.x * blockDim.x + threadIdx.x;   // B*S*H*DK/2 = 2^21
  int j = idx & 31;                                  // pair index, DK/2=32
  int s = (idx >> 9) & 2047;
  int b = idx >> 20;
  int h = (idx >> 5) & 15;
  size_t base = (((size_t)b * S_ + s) * H_ + h) * DK_ + 2 * j;
  float inv = powf(10000.f, -(float)(2 * j) * (1.f / (float)DK_));
  float ang = (float)s * inv;
  float sn, c;
  sincosf(ang, &sn, &c);
  float q1 = bf2f(q[base]), q2 = bf2f(q[base + 1]);
  q[base]     = f2bf(q1 * c - q2 * sn);
  q[base + 1] = f2bf(q1 * sn + q2 * c);
  float k1 = bf2f(k[base]), k2 = bf2f(k[base + 1]);
  k[base]     = f2bf(k1 * c - k2 * sn);
  k[base + 1] = f2bf(k1 * sn + k2 * c);
}

// ---------------------------------------------------------------- causal flash attention
// grid (S/128, B*H); block 256 = 8 waves; wave handles 16 q rows x DK=64.
// Iterate keys in 32-wide tiles: scores via 4 WMMA (2 ct x 2 kc), online softmax,
// P bounced through per-wave LDS to re-shape C-layout -> A-layout, PV via 4 WMMA
// against an LDS-transposed V tile (block-cooperatively staged).
#define LVT 40
__global__ void attn_kernel(const bf16* __restrict__ Q, const bf16* __restrict__ Kt,
                            const bf16* __restrict__ V, bf16* __restrict__ O) {
  __shared__ bf16 ldsVT[64 * LVT];        // [dk][key_local(32)+pad]
  __shared__ bf16 ldsP[8 * 16 * LVT];     // per-wave 16x32 P bounce

  const int tid  = threadIdx.x;
  const int wave = tid >> 5;
  const int lane = tid & 31;
  const int lh   = lane >> 4;
  const int ll   = lane & 15;
  const int bh   = blockIdx.y;
  const int b    = bh / H_;
  const int h    = bh % H_;
  const int qt   = blockIdx.x;
  const int m0   = qt * 128 + wave * 16;

  const bf16* Qb = Q + (size_t)b * S_ * D_ + h * DK_;   // (s,dk) at [s*D_ + dk]
  const bf16* Kb = Kt + (size_t)b * S_ * D_ + h * DK_;
  const bf16* Vb = V + (size_t)b * S_ * D_ + h * DK_;

  Frag qa[2];
#pragma unroll
  for (int kc = 0; kc < 2; ++kc) {
    const bf16* p = Qb + (size_t)(m0 + ll) * D_ + kc * 32 + lh * 16;
    qa[kc].lo = *reinterpret_cast<const uint4*>(p);
    qa[kc].hi = *reinterpret_cast<const uint4*>(p + 8);
  }

  float mrow[8], lrow[8];
#pragma unroll
  for (int r = 0; r < 8; ++r) { mrow[r] = -3.0e38f; lrow[r] = 0.f; }
  v8f o[4] = {vzero(), vzero(), vzero(), vzero()};

  const int nIter = qt * 4 + 4;           // covers keys [0, qt*128+128)
  for (int it = 0; it < nIter; ++it) {
    const int kb = it * 32;
    __syncthreads();
    {   // stage V[kb..kb+31][0..63] transposed into ldsVT[dk][key]
      int kr = tid >> 3;             // 0..31 key row
      int c8 = (tid & 7) * 8;        // dk chunk
      union { uint4 u; bf16 hv[8]; } t;
      t.u = *reinterpret_cast<const uint4*>(Vb + (size_t)(kb + kr) * D_ + c8);
#pragma unroll
      for (int j = 0; j < 8; ++j) ldsVT[(c8 + j) * LVT + kr] = t.hv[j];
    }
    __syncthreads();

    if (kb <= m0 + 15) {
      // ---- scores S = Q K^T / sqrt(DK), two 16x16 tiles over 32 keys
      v8f s[2] = {vzero(), vzero()};
#pragma unroll
      for (int ct = 0; ct < 2; ++ct) {
#pragma unroll
        for (int kc = 0; kc < 2; ++kc) {
          const bf16* p = Kb + (size_t)(kb + ct * 16 + ll) * D_ + kc * 32 + lh * 16;
          Frag bk_; bk_.lo = *reinterpret_cast<const uint4*>(p);
                    bk_.hi = *reinterpret_cast<const uint4*>(p + 8);
          s[ct] = wmma_bf16(qa[kc], bk_, s[ct]);
        }
      }
      // ---- scale + causal mask + row max
      const float scale = 0.125f;   // 1/sqrt(64)
      float rowmax[8];
#pragma unroll
      for (int r = 0; r < 8; ++r) rowmax[r] = -3.0e38f;
#pragma unroll
      for (int ct = 0; ct < 2; ++ct) {
        int kk = kb + ct * 16 + ll;
#pragma unroll
        for (int r = 0; r < 8; ++r) {
          int m = m0 + r + lh * 8;
          float sv = s[ct][r] * scale;
          sv = (kk <= m) ? sv : -3.0e38f;
          s[ct][r] = sv;
          rowmax[r] = fmaxf(rowmax[r], sv);
        }
      }
#pragma unroll
      for (int xm = 1; xm < 16; xm <<= 1)
#pragma unroll
        for (int r = 0; r < 8; ++r)
          rowmax[r] = fmaxf(rowmax[r], __shfl_xor(rowmax[r], xm, 32));
      // ---- online softmax update
      float corr[8], rowsum[8];
#pragma unroll
      for (int r = 0; r < 8; ++r) {
        float nm = fmaxf(mrow[r], rowmax[r]);
        corr[r] = __expf(mrow[r] - nm);
        mrow[r] = nm;
        rowsum[r] = 0.f;
      }
#pragma unroll
      for (int ct = 0; ct < 2; ++ct)
#pragma unroll
        for (int r = 0; r < 8; ++r) {
          float p = __expf(s[ct][r] - mrow[r]);
          s[ct][r] = p;
          rowsum[r] += p;
        }
#pragma unroll
      for (int xm = 1; xm < 16; xm <<= 1)
#pragma unroll
        for (int r = 0; r < 8; ++r)
          rowsum[r] += __shfl_xor(rowsum[r], xm, 32);
#pragma unroll
      for (int r = 0; r < 8; ++r) lrow[r] = lrow[r] * corr[r] + rowsum[r];
#pragma unroll
      for (int nt = 0; nt < 4; ++nt)
#pragma unroll
        for (int r = 0; r < 8; ++r) o[nt][r] *= corr[r];
      // ---- bounce P through LDS (C layout -> A layout), then PV
      bf16* pl = ldsP + wave * 16 * LVT;
#pragma unroll
      for (int ct = 0; ct < 2; ++ct)
#pragma unroll
        for (int r = 0; r < 8; ++r)
          pl[(r + lh * 8) * LVT + ct * 16 + ll] = f2bf(s[ct][r]);
      Frag pa;
      {
        const bf16* pb = pl + ll * LVT + lh * 16;
        pa.lo = *reinterpret_cast<const uint4*>(pb);
        pa.hi = *reinterpret_cast<const uint4*>(pb + 8);
      }
#pragma unroll
      for (int nt = 0; nt < 4; ++nt) {
        const bf16* vb = &ldsVT[(nt * 16 + ll) * LVT + lh * 16];
        Frag bv_; bv_.lo = *reinterpret_cast<const uint4*>(vb);
                  bv_.hi = *reinterpret_cast<const uint4*>(vb + 8);
        o[nt] = wmma_bf16(pa, bv_, o[nt]);
      }
    }
  }

  // ---- write ctx [B,S,H,DK] in bf16
#pragma unroll
  for (int nt = 0; nt < 4; ++nt)
#pragma unroll
    for (int r = 0; r < 8; ++r) {
      int m = m0 + r + lh * 8;
      int dk = nt * 16 + ll;
      O[(size_t)b * S_ * D_ + (size_t)m * D_ + h * DK_ + dk] = f2bf(o[nt][r] / lrow[r]);
    }
}

// ---------------------------------------------------------------- host launch
extern "C" void kernel_launch(void* const* d_in, const int* in_sizes, int n_in,
                              void* d_out, int out_size, void* d_ws, size_t ws_size,
                              hipStream_t stream) {
  (void)in_sizes; (void)n_in; (void)out_size; (void)ws_size;
  const float* x   = (const float*)d_in[0];
  // d_in[1] = mask (causal; handled analytically)
  const float* Wq  = (const float*)d_in[2];
  const float* bq  = (const float*)d_in[3];
  const float* Wk  = (const float*)d_in[4];
  const float* bk  = (const float*)d_in[5];
  const float* Wv  = (const float*)d_in[6];
  const float* bv  = (const float*)d_in[7];
  const float* Wo  = (const float*)d_in[8];
  const float* bo  = (const float*)d_in[9];
  const float* w1  = (const float*)d_in[10];
  const float* w2  = (const float*)d_in[11];
  const float* w3  = (const float*)d_in[12];
  const float* n1w = (const float*)d_in[13];
  const float* n2w = (const float*)d_in[14];
  float* out = (float*)d_out;

  char* ws = (char*)d_ws;
  size_t off = 0;
  auto alloc = [&](size_t bytes) -> char* {
    char* p = ws + off; off += (bytes + 255) & ~(size_t)255; return p;
  };
  bf16* wq_b = (bf16*)alloc((size_t)D_ * D_ * 2);
  bf16* wk_b = (bf16*)alloc((size_t)D_ * D_ * 2);
  bf16* wv_b = (bf16*)alloc((size_t)D_ * D_ * 2);
  bf16* wo_b = (bf16*)alloc((size_t)D_ * D_ * 2);
  bf16* w1_b = (bf16*)alloc((size_t)DF_ * D_ * 2);
  bf16* w3_b = (bf16*)alloc((size_t)DF_ * D_ * 2);
  bf16* w2_b = (bf16*)alloc((size_t)D_ * DF_ * 2);
  bf16* h_b  = (bf16*)alloc((size_t)M_ * D_ * 2);
  bf16* q_b  = (bf16*)alloc((size_t)M_ * D_ * 2);
  bf16* k_b  = (bf16*)alloc((size_t)M_ * D_ * 2);
  bf16* v_b  = (bf16*)alloc((size_t)M_ * D_ * 2);
  bf16* ctx_b= (bf16*)alloc((size_t)M_ * D_ * 2);
  float* x1  = (float*)alloc((size_t)M_ * D_ * 4);
  bf16* h2_b = (bf16*)alloc((size_t)M_ * D_ * 2);
  bf16* ff_b = (bf16*)alloc((size_t)M_ * DF_ * 2);

  auto cvt = [&](const float* s, bf16* d, size_t n) {
    int n4 = (int)(n / 4);
    cvt_f32_to_bf16_kernel<<<(n4 + 255) / 256, 256, 0, stream>>>(s, d, n4);
  };
  cvt(Wq, wq_b, (size_t)D_ * D_);
  cvt(Wk, wk_b, (size_t)D_ * D_);
  cvt(Wv, wv_b, (size_t)D_ * D_);
  cvt(Wo, wo_b, (size_t)D_ * D_);
  cvt(w1, w1_b, (size_t)DF_ * D_);
  cvt(w3, w3_b, (size_t)DF_ * D_);
  cvt(w2, w2_b, (size_t)D_ * DF_);

  // 1) h = rmsnorm(x, n1w)  (bf16)
  rmsnorm_bf16_kernel<<<M_, 256, 0, stream>>>(x, n1w, h_b);

  // 2) q,k,v = h @ W{q,k,v}^T + b   (bf16 out)
  dim3 gq(D_ / 64, M_ / 128);
  gemm_wmma_kernel<0><<<gq, 256, 0, stream>>>(h_b, wq_b, nullptr, bq, nullptr, nullptr, q_b, D_, D_);
  gemm_wmma_kernel<0><<<gq, 256, 0, stream>>>(h_b, wk_b, nullptr, bk, nullptr, nullptr, k_b, D_, D_);
  gemm_wmma_kernel<0><<<gq, 256, 0, stream>>>(h_b, wv_b, nullptr, bv, nullptr, nullptr, v_b, D_, D_);

  // 3) RoPE on q,k (in place; q,k fully rewritten by step 2 each launch)
  int npair = B_ * S_ * H_ * (DK_ / 2);
  rope_kernel<<<npair / 256, 256, 0, stream>>>(q_b, k_b);

  // 4) causal flash attention -> ctx (bf16, [B,S,H*DK] == [M,D])
  dim3 ga(S_ / 128, B_ * H_);
  attn_kernel<<<ga, 256, 0, stream>>>(q_b, k_b, v_b, ctx_b);

  // 5) x1 = ctx @ Wo^T + bo + x   (f32)
  gemm_wmma_kernel<0><<<gq, 256, 0, stream>>>(ctx_b, wo_b, nullptr, bo, x, x1, nullptr, D_, D_);

  // 6) h2 = rmsnorm(x1, n2w)
  rmsnorm_bf16_kernel<<<M_, 256, 0, stream>>>(x1, n2w, h2_b);

  // 7) ff = gelu(h2 @ w1^T) * (h2 @ w3^T)   (bf16, gated fused)
  dim3 gf(DF_ / 64, M_ / 128);
  gemm_wmma_kernel<1><<<gf, 256, 0, stream>>>(h2_b, w1_b, w3_b, nullptr, nullptr, nullptr, ff_b, DF_, D_);

  // 8) out = ff @ w2^T + x1   (f32)
  gemm_wmma_kernel<0><<<gq, 256, 0, stream>>>(ff_b, w2_b, nullptr, nullptr, x1, out, nullptr, D_, DF_);
}
